// MultiHeadSelfAttention_11201274708177
// MI455X (gfx1250) — matrix operand order repaired and checked by reading, not compiled, not run on hardware
//
#include <hip/hip_runtime.h>

typedef _Float16 f16;
typedef __attribute__((ext_vector_type(16))) _Float16 v16h;
typedef __attribute__((ext_vector_type(8)))  _Float16 v8h;
typedef __attribute__((ext_vector_type(4)))  _Float16 v4h;
typedef __attribute__((ext_vector_type(8)))  float    v8f;

#define EMBED   1024
#define SEQ     2048
#define BATCH   2
#define HEADS   16
#define HDIM    64

__device__ __forceinline__ v16h cat16(v8h lo, v8h hi) {
  return __builtin_shufflevector(lo, hi, 0,1,2,3,4,5,6,7,8,9,10,11,12,13,14,15);
}

__device__ __forceinline__ v8f wmma_f16(v16h a, v16h b, v8f c) {
  // v_wmma_f32_16x16x32_f16: (neg_a, A, neg_b, B, c_mod, C, reuse_a, reuse_b)
  return __builtin_amdgcn_wmma_f32_16x16x32_f16(false, a, false, b, (short)0, c, false, false);
}

// ---------------------------------------------------------------------------
// fp32 -> fp16 conversion (vectorized x4)
// ---------------------------------------------------------------------------
__global__ void cvt_f16_kernel(const float* __restrict__ src, f16* __restrict__ dst, int n4) {
  int i = blockIdx.x * blockDim.x + threadIdx.x;
  if (i < n4) {
    float4 v = ((const float4*)src)[i];
    v4h o;
    o[0] = (f16)v.x; o[1] = (f16)v.y; o[2] = (f16)v.z; o[3] = (f16)v.w;
    ((v4h*)dst)[i] = o;
  }
}

// ---------------------------------------------------------------------------
// WMMA GEMM:  C[m,o] = sum_k A[m,k] * W[o,k]   (A:[M,K] f16, W:[O,K] f16)
// MODE 0: scatter f16 results into Q/K/V buffers laid out [B,H,N,D]
// MODE 1: outF[m*O+o] = acc + bias[o]  (fp32)
// Block: 256 threads = 8 waves; tile 128(M) x 128(O); BK = 32.
// Waves arranged 4(row-slabs of 32) x 2(col-slabs of 64); each wave does a
// 32x64 sub-tile = 2x4 WMMA fragments -> 8 WMMAs per k-step per wave.
// Next k-step's tiles are prefetched (global_prefetch_b8) during compute.
// ---------------------------------------------------------------------------
template<int MODE>
__global__ __launch_bounds__(256) void gemm_wmma_kernel(
    const f16* __restrict__ A, const f16* __restrict__ W,
    const float* __restrict__ bias, float* __restrict__ outF,
    f16* __restrict__ Qp, f16* __restrict__ Kp, f16* __restrict__ Vp,
    int M, int Kdim, int O)
{
  __shared__ f16 As[128][40];   // row stride 80B (mult of 16B, conflict-spread)
  __shared__ f16 Bs[128][40];

  const int tid  = threadIdx.x;
  const int wave = tid >> 5;
  const int lane = tid & 31;
  const int l16  = lane >> 4;
  const int ln   = lane & 15;
  const int m0   = blockIdx.y * 128;
  const int o0   = blockIdx.x * 128;
  const int mw   = (wave & 3) * 32;    // row slab within tile
  const int ow   = (wave >> 2) * 64;   // col slab within tile

  v8f acc[2][4];
  #pragma unroll
  for (int i = 0; i < 2; ++i)
    #pragma unroll
    for (int j = 0; j < 4; ++j)
      #pragma unroll
      for (int r = 0; r < 8; ++r) acc[i][j][r] = 0.f;

  const int lrow  = tid >> 1;          // 0..127
  const int lhalf = (tid & 1) * 16;    // 0 or 16 (f16 elements)

  for (int k0 = 0; k0 < Kdim; k0 += 32) {
    // --- cooperative stage of A and B tiles into LDS (32B per thread each) ---
    const f16* ag = A + (size_t)(m0 + lrow) * Kdim + k0 + lhalf;
    *(v8h*)&As[lrow][lhalf]     = *(const v8h*)(ag);
    *(v8h*)&As[lrow][lhalf + 8] = *(const v8h*)(ag + 8);
    const f16* wg = W + (size_t)(o0 + lrow) * Kdim + k0 + lhalf;
    *(v8h*)&Bs[lrow][lhalf]     = *(const v8h*)(wg);
    *(v8h*)&Bs[lrow][lhalf + 8] = *(const v8h*)(wg + 8);

    // --- prefetch next k-step's tiles into cache (global_prefetch_b8) ---
    if (k0 + 32 < Kdim) {
      __builtin_prefetch(ag + 32, 0, 3);
      __builtin_prefetch(wg + 32, 0, 3);
    }
    __syncthreads();

    // --- A fragments (16x32): per-lane k-runs [l16*8,+8) and [16+l16*8,+8) ---
    v16h afr[2];
    #pragma unroll
    for (int i = 0; i < 2; ++i)
      afr[i] = cat16(*(const v8h*)&As[mw + i * 16 + ln][l16 * 8],
                     *(const v8h*)&As[mw + i * 16 + ln][16 + l16 * 8]);
    // --- B fragments (32x16): per-lane k-runs [l16*8,+8) and [16+l16*8,+8) ---
    v16h bfr[4];
    #pragma unroll
    for (int j = 0; j < 4; ++j)
      bfr[j] = cat16(*(const v8h*)&Bs[ow + j * 16 + ln][l16 * 8],
                     *(const v8h*)&Bs[ow + j * 16 + ln][16 + l16 * 8]);

    #pragma unroll
    for (int i = 0; i < 2; ++i)
      #pragma unroll
      for (int j = 0; j < 4; ++j)
        acc[i][j] = wmma_f16(afr[i], bfr[j], acc[i][j]);
    __syncthreads();
  }

  // --- epilogue: C/D layout row = r + 8*l16, col = ln ---
  #pragma unroll
  for (int i = 0; i < 2; ++i) {
    #pragma unroll
    for (int j = 0; j < 4; ++j) {
      const v8f& cf = acc[i][j];
      const int og = o0 + ow + j * 16 + ln;
      if (MODE == 1) {
        const float bv = bias[og];
        #pragma unroll
        for (int r = 0; r < 8; ++r) {
          const int mg = m0 + mw + i * 16 + r + 8 * l16;
          outF[(size_t)mg * O + og] = cf[r] + bv;
        }
      } else {
        const int s   = og >> 10;          // 0:Q 1:K 2:V
        const int rem = og & 1023;
        const int h   = rem >> 6;
        const int d   = rem & 63;
        f16* dst = (s == 0) ? Qp : ((s == 1) ? Kp : Vp);
        #pragma unroll
        for (int r = 0; r < 8; ++r) {
          const int mg = m0 + mw + i * 16 + r + 8 * l16;
          const int b  = mg >> 11;         // / SEQ
          const int nn = mg & 2047;
          dst[(((size_t)(b * HEADS + h)) * SEQ + nn) * HDIM + d] = (f16)cf[r];
        }
      }
    }
  }
}

// ---------------------------------------------------------------------------
// Flash attention: per (b,h) and 64-query tile. 128 threads = 4 waves, each
// wave owns 16 query rows. Keys processed in chunks of 64 with online
// softmax; row max via cross-lane shuffles, row sum via WMMA against an
// all-ones B matrix. Next K/V tiles prefetched during compute.
// Output written f16 into [B,N,C] layout for the projection GEMM.
// ---------------------------------------------------------------------------
__global__ __launch_bounds__(128) void attn_wmma_kernel(
    const f16* __restrict__ Qg, const f16* __restrict__ Kg,
    const f16* __restrict__ Vg, f16* __restrict__ Og)
{
  __shared__ f16 kt[64][72];        // K tile [key][d],            stride 144B
  __shared__ f16 vt[64][72];        // V tile transposed [d][key], stride 144B
  __shared__ f16 pbuf[4][16][72];   // per-wave P staging [row][key]

  const int tid  = threadIdx.x;
  const int wave = tid >> 5;
  const int lane = tid & 31;
  const int l16  = lane >> 4;
  const int ln   = lane & 15;
  const int bh   = blockIdx.y;             // b*HEADS + h
  const int q0   = blockIdx.x * 64;
  const float scale = 0.125f;              // HDIM^-0.5

  const f16* Qb = Qg + (size_t)bh * SEQ * HDIM;
  const f16* Kb = Kg + (size_t)bh * SEQ * HDIM;
  const f16* Vb = Vg + (size_t)bh * SEQ * HDIM;

  // Q A-fragments for this wave's 16 rows, two d-chunks of 32
  const int qrow = q0 + wave * 16 + ln;
  v16h qa[2];
  #pragma unroll
  for (int dc = 0; dc < 2; ++dc) {
    const f16* qp = Qb + (size_t)qrow * HDIM + dc * 32;
    qa[dc] = cat16(*(const v8h*)(qp + l16 * 8),
                   *(const v8h*)(qp + 16 + l16 * 8));
  }

  // all-ones B fragment (32x16 of 1.0) for WMMA row-sum
  v16h ones;
  #pragma unroll
  for (int i = 0; i < 16; ++i) ones[i] = (f16)1.0f;

  float mrun[8], lrun[8];
  v8f o[4];
  #pragma unroll
  for (int r = 0; r < 8; ++r) {
    mrun[r] = -1e30f; lrun[r] = 0.f;
    o[0][r] = 0.f; o[1][r] = 0.f; o[2][r] = 0.f; o[3][r] = 0.f;
  }

  const int crow  = tid >> 1;         // 0..63 : key row for staging
  const int cpart = (tid & 1) * 32;   // d-offset for staging (0 or 32)

  for (int kc = 0; kc < SEQ; kc += 64) {
    // --- stage K tile (row-major) and V tile (transposed) into LDS ---
    {
      const f16* kp = Kb + (size_t)(kc + crow) * HDIM + cpart;
      *(v8h*)&kt[crow][cpart]      = *(const v8h*)kp;
      *(v8h*)&kt[crow][cpart + 8]  = *(const v8h*)(kp + 8);
      *(v8h*)&kt[crow][cpart + 16] = *(const v8h*)(kp + 16);
      *(v8h*)&kt[crow][cpart + 24] = *(const v8h*)(kp + 24);
      const f16* vp = Vb + (size_t)(kc + crow) * HDIM + cpart;
      #pragma unroll
      for (int q = 0; q < 4; ++q) {
        v8h vv = *(const v8h*)(vp + q * 8);
        #pragma unroll
        for (int i = 0; i < 8; ++i)
          vt[cpart + q * 8 + i][crow] = vv[i];
      }
      // --- prefetch next key chunk's K/V rows (global_prefetch_b8) ---
      if (kc + 64 < SEQ) {
        __builtin_prefetch(kp + (size_t)64 * HDIM, 0, 3);
        __builtin_prefetch(vp + (size_t)64 * HDIM, 0, 3);
      }
    }
    __syncthreads();

    // --- S = Q * K^T for 64 keys: four 16x16 f32 fragments ---
    v8f s[4];
    #pragma unroll
    for (int j = 0; j < 4; ++j)
      #pragma unroll
      for (int r = 0; r < 8; ++r) s[j][r] = 0.f;
    #pragma unroll
    for (int dc = 0; dc < 2; ++dc) {
      #pragma unroll
      for (int j = 0; j < 4; ++j) {
        v16h kb = cat16(*(const v8h*)&kt[j * 16 + ln][dc * 32 + l16 * 8],
                        *(const v8h*)&kt[j * 16 + ln][dc * 32 + 16 + l16 * 8]);
        s[j] = wmma_f16(qa[dc], kb, s[j]);
      }
    }

    // --- online softmax (rows: r + 8*l16; key cols across 16-lane group) ---
    #pragma unroll
    for (int r = 0; r < 8; ++r) {
      float v0 = s[0][r] * scale;
      float v1 = s[1][r] * scale;
      float v2 = s[2][r] * scale;
      float v3 = s[3][r] * scale;
      float mx = fmaxf(fmaxf(v0, v1), fmaxf(v2, v3));
      #pragma unroll
      for (int off = 1; off < 16; off <<= 1)
        mx = fmaxf(mx, __shfl_xor(mx, off, 32));
      const float nm    = fmaxf(mrun[r], mx);
      const float alpha = __expf(mrun[r] - nm);
      mrun[r] = nm;
      lrun[r] *= alpha;
      o[0][r] *= alpha; o[1][r] *= alpha; o[2][r] *= alpha; o[3][r] *= alpha;
      const int m = r + 8 * l16;
      pbuf[wave][m][ln]      = (f16)__expf(v0 - nm);
      pbuf[wave][m][16 + ln] = (f16)__expf(v1 - nm);
      pbuf[wave][m][32 + ln] = (f16)__expf(v2 - nm);
      pbuf[wave][m][48 + ln] = (f16)__expf(v3 - nm);
    }

    // --- reload P as two A-fragments (per-wave LDS, dscnt-ordered) ---
    v16h pa0 = cat16(*(const v8h*)&pbuf[wave][ln][l16 * 8],
                     *(const v8h*)&pbuf[wave][ln][16 + l16 * 8]);
    v16h pa1 = cat16(*(const v8h*)&pbuf[wave][ln][32 + l16 * 8],
                     *(const v8h*)&pbuf[wave][ln][48 + l16 * 8]);

    // --- row sums via WMMA against all-ones B (broadcasts into all lanes) ---
    v8f ls;
    #pragma unroll
    for (int r = 0; r < 8; ++r) ls[r] = 0.f;
    ls = wmma_f16(pa0, ones, ls);
    ls = wmma_f16(pa1, ones, ls);
    #pragma unroll
    for (int r = 0; r < 8; ++r) lrun[r] += ls[r];

    // --- O += P * V  (4 d-chunks of 16, 2 key-halves of 32) ---
    #pragma unroll
    for (int c = 0; c < 4; ++c) {
      v16h vb0 = cat16(*(const v8h*)&vt[c * 16 + ln][l16 * 8],
                       *(const v8h*)&vt[c * 16 + ln][16 + l16 * 8]);
      o[c] = wmma_f16(pa0, vb0, o[c]);
      v16h vb1 = cat16(*(const v8h*)&vt[c * 16 + ln][32 + l16 * 8],
                       *(const v8h*)&vt[c * 16 + ln][48 + l16 * 8]);
      o[c] = wmma_f16(pa1, vb1, o[c]);
    }
    __syncthreads();
  }

  // --- finalize: divide by row sums, write [B,N,C] f16 ---
  const int b = bh >> 4;
  const int h = bh & 15;
  #pragma unroll
  for (int r = 0; r < 8; ++r) {
    const int m  = r + 8 * l16;
    const int qg = q0 + wave * 16 + m;
    const float inv = 1.0f / lrun[r];
    const size_t base = ((size_t)(b * SEQ + qg)) * EMBED + h * HDIM;
    #pragma unroll
    for (int c = 0; c < 4; ++c)
      Og[base + c * 16 + ln] = (f16)(o[c][r] * inv);
  }
}

// ---------------------------------------------------------------------------
extern "C" void kernel_launch(void* const* d_in, const int* in_sizes, int n_in,
                              void* d_out, int out_size, void* d_ws, size_t ws_size,
                              hipStream_t stream) {
  const float* x      = (const float*)d_in[0];   // [B,N,C]
  const float* w_qkv  = (const float*)d_in[1];   // [3C,C]
  const float* w_proj = (const float*)d_in[2];   // [C,C]
  const float* b_proj = (const float*)d_in[3];   // [C]
  float* out = (float*)d_out;

  char* ws = (char*)d_ws;
  f16* Xh    = (f16*)(ws + 0);             //  8 MB  [4096,1024]
  f16* Wqkvh = (f16*)(ws + 8388608);       //  6 MB  [3072,1024]
  f16* Wph   = (f16*)(ws + 14680064);      //  2 MB  [1024,1024]
  f16* Qh    = (f16*)(ws + 16777216);      //  8 MB  [B,H,N,D]
  f16* Kh    = (f16*)(ws + 25165824);      //  8 MB
  f16* Vh    = (f16*)(ws + 33554432);      //  8 MB
  f16* Oh    = (f16*)(ws + 41943040);      //  8 MB  [4096,1024]

  // fp32 -> fp16 conversions
  cvt_f16_kernel<<<(4194304 / 4 + 255) / 256, 256, 0, stream>>>(x,      Xh,    4194304 / 4);
  cvt_f16_kernel<<<(3145728 / 4 + 255) / 256, 256, 0, stream>>>(w_qkv,  Wqkvh, 3145728 / 4);
  cvt_f16_kernel<<<(1048576 / 4 + 255) / 256, 256, 0, stream>>>(w_proj, Wph,   1048576 / 4);

  // QKV projection: [4096,1024] x [3072,1024]^T -> scattered Q/K/V
  gemm_wmma_kernel<0><<<dim3(3072 / 128, 4096 / 128), 256, 0, stream>>>(
      Xh, Wqkvh, nullptr, nullptr, Qh, Kh, Vh, 4096, 1024, 3072);

  // attention: 32 q-tiles x 32 (b,h) pairs
  attn_wmma_kernel<<<dim3(SEQ / 64, BATCH * HEADS), 128, 0, stream>>>(Qh, Kh, Vh, Oh);

  // output projection + bias -> fp32 d_out
  gemm_wmma_kernel<1><<<dim3(1024 / 128, 4096 / 128), 256, 0, stream>>>(
      Oh, Wph, b_proj, nullptr, nullptr, nullptr, nullptr, 4096, 1024, 1024);
}
